// LayoutVAE_72103910965765
// MI455X (gfx1250) — matrix operand order repaired and checked by reading, not compiled, not run on hardware
//
#include <hip/hip_runtime.h>

#define NOBJ      10000
#define NOBJ_PAD  10048
#define NTRI      160000
#define DOBJ      128
#define DPRED     256
#define HDIM      512
#define DOUTD     128

typedef __attribute__((ext_vector_type(16))) __bf16        v16bf;
typedef __attribute__((ext_vector_type(8)))  float         v8f;
typedef __attribute__((ext_vector_type(8)))  unsigned short us8;
typedef __attribute__((ext_vector_type(4)))  unsigned short us4;

// ---- helpers ---------------------------------------------------------------

static __device__ __forceinline__ unsigned short f2bf(float x) {
  unsigned int u = __float_as_uint(x);
  u += 0x7FFFu + ((u >> 16) & 1u);          // round-to-nearest-even
  return (unsigned short)(u >> 16);
}

static __device__ __forceinline__ v8f wmma_bf16(v16bf a, v16bf b, v8f c) {
  // emits v_wmma_f32_16x16x32_bf16
  return __builtin_amdgcn_wmma_f32_16x16x32_bf16(false, a, false, b, (short)0, c,
                                                 false, false);
}

// A fragment (16x32 bf16) from LDS row-major tile.
// ISA A-map: lane<16 holds K {kt*32+0..7, kt*32+16..23}; lane>=16 adds 8.
static __device__ __forceinline__ v16bf afrag(const unsigned short* lds, int stride,
                                              int mt, int kt, int lane) {
  const unsigned short* p =
      lds + (mt * 16 + (lane & 15)) * stride + kt * 32 + ((lane & 16) >> 1);
  union { us8 h[2]; v16bf v; } u;
  u.h[0] = *(const us8*)(p);
  u.h[1] = *(const us8*)(p + 16);
  return u.v;
}

// B fragment (32x16 bf16) from pre-swizzled weights: frag f = nt*KT + kt.
static __device__ __forceinline__ v16bf bfrag(const unsigned short* wb, int f, int lane) {
  const unsigned short* p = wb + ((size_t)f << 9) + (lane << 4);
  union { us8 h[2]; v16bf v; } u;
  u.h[0] = *(const us8*)(p);
  u.h[1] = *(const us8*)(p + 8);
  return u.v;
}

static __device__ __forceinline__ void st4bf(unsigned short* d, float4 v, float s) {
  us4 h;
  h.x = f2bf(v.x * s); h.y = f2bf(v.y * s); h.z = f2bf(v.z * s); h.w = f2bf(v.w * s);
  *(us4*)d = h;
}

static __device__ __forceinline__ v8f bcast8(float v) {
  v8f r;
#pragma unroll
  for (int j = 0; j < 8; ++j) r[j] = v;
  return r;
}

// ---- kernel: zero workspace ------------------------------------------------

__global__ __launch_bounds__(256) void zero_kernel(float* p, int n) {
  int i = blockIdx.x * 256 + threadIdx.x;
  if (i < n) p[i] = 0.0f;
}

// ---- kernel: weight f32 -> bf16 B-fragment swizzle -------------------------
// W is Kdim x Ndim row-major (fan_in x fan_out). dst frag layout:
// dst[f*512 + lane*16 + e] = W[(kt*32 + (e&7) + ((e&8)<<1) + ((lane&16)>>1)) * Ndim + nt*16 + (lane&15)]

__global__ __launch_bounds__(256) void prep_kernel(const float* __restrict__ W,
                                                   unsigned short* __restrict__ dst,
                                                   int Kdim, int Ndim) {
  int KT = Kdim >> 5;
  int total = Kdim * Ndim;
  int tid = blockIdx.x * 256 + threadIdx.x;
  if (tid >= total) return;
  int f = tid >> 9, r = tid & 511;
  int lane = r >> 4, e = r & 15;
  int nt = f / KT, kt = f - nt * KT;
  int k = kt * 32 + (e & 7) + ((e & 8) << 1) + ((lane & 16) >> 1);
  int n = nt * 16 + (lane & 15);
  dst[tid] = f2bf(W[(size_t)k * Ndim + n]);
}

// ---- kernel: edge degree counts --------------------------------------------

__global__ __launch_bounds__(256) void count_kernel(const int* __restrict__ edges,
                                                    float* __restrict__ counts) {
  int i = blockIdx.x * 256 + threadIdx.x;
  if (i < NTRI) {
    atomicAdd(counts + edges[2 * (size_t)i + 0], 1.0f);
    atomicAdd(counts + edges[2 * (size_t)i + 1], 1.0f);
  }
}

// ---- scatter of one GEMM2 output tile --------------------------------------

static __device__ __forceinline__ void scatter_tile(
    v8f acc, int nt, int mt, int row0, int Nc, int hi8,
    const int* __restrict__ sidx, const int* __restrict__ oidx,
    float* __restrict__ pooled, float* __restrict__ outp) {
  if (nt < 32) {                        // new_s -> pooled[s_idx]
    int col = nt * 16 + Nc;
#pragma unroll
    for (int j = 0; j < 8; ++j) {
      int rl = mt * 16 + j + hi8;
      atomicAdd(pooled + (size_t)sidx[rl] * HDIM + col, fmaxf(acc[j], 0.0f));
    }
  } else if (nt < 40) {                 // new_p core -> d_out
    int col = nt * 16 + Nc - 512;
#pragma unroll
    for (int j = 0; j < 8; ++j) {
      int rl = mt * 16 + j + hi8;
      outp[(size_t)(row0 + rl) * DOUTD + col] = fmaxf(acc[j], 0.0f);
    }
  } else {                              // new_o -> pooled[o_idx]
    int col = nt * 16 + Nc - 640;
#pragma unroll
    for (int j = 0; j < 8; ++j) {
      int rl = mt * 16 + j + hi8;
      atomicAdd(pooled + (size_t)oidx[rl] * HDIM + col, fmaxf(acc[j], 0.0f));
    }
  }
}

// ---- kernel: fused triples pipeline ----------------------------------------
// 64 rows/block. LDS: curt[64*512]bf16, ht[64*512]bf16, sidx[64], oidx[64].

__global__ __launch_bounds__(256) void triples_kernel(
    const float* __restrict__ obj, const float* __restrict__ pred,
    const int* __restrict__ edges, const float* __restrict__ b1,
    const float* __restrict__ b2, const unsigned short* __restrict__ w1b,
    const unsigned short* __restrict__ w2b, float* __restrict__ pooled,
    float* __restrict__ outp) {
  extern __shared__ __align__(16) char smem[];
  unsigned short* curt = (unsigned short*)smem;
  unsigned short* ht   = curt + 64 * 512;
  int* sidx = (int*)(ht + 64 * 512);
  int* oidx = sidx + 64;

  int tid  = threadIdx.x;
  int row0 = blockIdx.x * 64;

  if (tid < 64) {
    sidx[tid] = edges[2 * (size_t)(row0 + tid) + 0];
    oidx[tid] = edges[2 * (size_t)(row0 + tid) + 1];
  }
  // Stage cur_t = [obj[s] | pred | obj[o]] as bf16. 4 threads/row, 128 cols each.
  {
    int srow  = tid >> 2;
    int cbase = (tid & 3) << 7;
    int gs = edges[2 * (size_t)(row0 + srow) + 0];
    int go = edges[2 * (size_t)(row0 + srow) + 1];
    const float* src;
    if (cbase == 0)        src = obj + (size_t)gs * DOBJ;
    else if (cbase == 384) src = obj + (size_t)go * DOBJ;
    else                   src = pred + (size_t)(row0 + srow) * DPRED + (cbase - 128);
    unsigned short* drow = curt + srow * 512 + cbase;
#pragma unroll
    for (int i = 0; i < 32; ++i)
      st4bf(drow + i * 4, *(const float4*)(src + i * 4), 1.0f);
  }
  __syncthreads();

  int w = tid >> 5, lane = tid & 31;
  int Nc  = lane & 15;
  int hi8 = (lane & 16) >> 1;

  // GEMM1: h = relu(cur_t @ w1 + b1). 2x2 register blocking: 2mtb x 16ntb = 32.
  for (int tb = w; tb < 32; tb += 8) {
    int nt0 = (tb & 15) * 2, mt0 = (tb >> 4) * 2;
    v8f a00 = bcast8(b1[nt0 * 16 + Nc]);
    v8f a01 = bcast8(b1[(nt0 + 1) * 16 + Nc]);
    v8f a10 = a00, a11 = a01;
#pragma unroll
    for (int kt = 0; kt < 16; ++kt) {
      v16bf av0 = afrag(curt, 512, mt0,     kt, lane);
      v16bf av1 = afrag(curt, 512, mt0 + 1, kt, lane);
      v16bf bv0 = bfrag(w1b, nt0 * 16 + kt, lane);
      v16bf bv1 = bfrag(w1b, (nt0 + 1) * 16 + kt, lane);
      a00 = wmma_bf16(av0, bv0, a00);
      a01 = wmma_bf16(av0, bv1, a01);
      a10 = wmma_bf16(av1, bv0, a10);
      a11 = wmma_bf16(av1, bv1, a11);
    }
#pragma unroll
    for (int j = 0; j < 8; ++j) {
      int r0 = (mt0 * 16 + j + hi8) * 512, r1 = ((mt0 + 1) * 16 + j + hi8) * 512;
      ht[r0 + nt0 * 16 + Nc]        = f2bf(fmaxf(a00[j], 0.0f));
      ht[r0 + (nt0 + 1) * 16 + Nc]  = f2bf(fmaxf(a01[j], 0.0f));
      ht[r1 + nt0 * 16 + Nc]        = f2bf(fmaxf(a10[j], 0.0f));
      ht[r1 + (nt0 + 1) * 16 + Nc]  = f2bf(fmaxf(a11[j], 0.0f));
    }
  }
  __syncthreads();

  // GEMM2: new_t = relu(h @ w2 + b2), fused scatter. 2mtb x 36ntb = 72 blocks.
  for (int tb = w; tb < 72; tb += 8) {
    int ntb = tb % 36, mtb = tb / 36;
    int nt0 = ntb * 2, mt0 = mtb * 2;
    v8f a00 = bcast8(b2[nt0 * 16 + Nc]);
    v8f a01 = bcast8(b2[(nt0 + 1) * 16 + Nc]);
    v8f a10 = a00, a11 = a01;
#pragma unroll
    for (int kt = 0; kt < 16; ++kt) {
      v16bf av0 = afrag(ht, 512, mt0,     kt, lane);
      v16bf av1 = afrag(ht, 512, mt0 + 1, kt, lane);
      v16bf bv0 = bfrag(w2b, nt0 * 16 + kt, lane);
      v16bf bv1 = bfrag(w2b, (nt0 + 1) * 16 + kt, lane);
      a00 = wmma_bf16(av0, bv0, a00);
      a01 = wmma_bf16(av0, bv1, a01);
      a10 = wmma_bf16(av1, bv0, a10);
      a11 = wmma_bf16(av1, bv1, a11);
    }
    scatter_tile(a00, nt0,     mt0,     row0, Nc, hi8, sidx, oidx, pooled, outp);
    scatter_tile(a01, nt0 + 1, mt0,     row0, Nc, hi8, sidx, oidx, pooled, outp);
    scatter_tile(a10, nt0,     mt0 + 1, row0, Nc, hi8, sidx, oidx, pooled, outp);
    scatter_tile(a11, nt0 + 1, mt0 + 1, row0, Nc, hi8, sidx, oidx, pooled, outp);
  }
}

// ---- kernel: object pathway ------------------------------------------------
// new_obj = relu(relu(pooled_n @ u1 + c1) @ u2 + c2) + obj @ pw + pb

__global__ __launch_bounds__(256) void obj_kernel(
    const float* __restrict__ obj, const float* __restrict__ c1v,
    const float* __restrict__ c2v, const float* __restrict__ pbv,
    const unsigned short* __restrict__ u1b, const unsigned short* __restrict__ u2b,
    const unsigned short* __restrict__ pwb, const float* __restrict__ pooled,
    const float* __restrict__ counts, float* __restrict__ outo) {
  extern __shared__ __align__(16) char smem[];
  unsigned short* pt = (unsigned short*)smem;   // 64x512
  unsigned short* h2 = pt + 64 * 512;           // 64x512
  unsigned short* ot = h2 + 64 * 512;           // 64x128

  int tid  = threadIdx.x;
  int row0 = blockIdx.x * 64;
  int srow = tid >> 2;
  int grow = row0 + srow;                       // < NOBJ_PAD (padded ws)
  bool valid = grow < NOBJ;

  // stage normalized pooled (64x512) and obj residual tile (64x128)
  {
    float inv = 1.0f / fmaxf(counts[grow], 1.0f);
    int cbase = (tid & 3) << 7;
    const float* src = pooled + (size_t)grow * HDIM + cbase;
    unsigned short* drow = pt + srow * 512 + cbase;
#pragma unroll
    for (int i = 0; i < 32; ++i)
      st4bf(drow + i * 4, *(const float4*)(src + i * 4), inv);

    int ob = (tid & 3) << 5;
    const float* osrc = obj + (size_t)grow * DOBJ + ob;
    unsigned short* od = ot + srow * 128 + ob;
    float4 z; z.x = z.y = z.z = z.w = 0.0f;
#pragma unroll
    for (int i = 0; i < 8; ++i)
      st4bf(od + i * 4, valid ? *(const float4*)(osrc + i * 4) : z, 1.0f);
  }
  __syncthreads();

  int w = tid >> 5, lane = tid & 31;
  int Nc  = lane & 15;
  int hi8 = (lane & 16) >> 1;

  // h2 = relu(pooled_n @ u1 + c1). 2x2 blocking, 32 tile-blocks.
  for (int tb = w; tb < 32; tb += 8) {
    int nt0 = (tb & 15) * 2, mt0 = (tb >> 4) * 2;
    v8f a00 = bcast8(c1v[nt0 * 16 + Nc]);
    v8f a01 = bcast8(c1v[(nt0 + 1) * 16 + Nc]);
    v8f a10 = a00, a11 = a01;
#pragma unroll
    for (int kt = 0; kt < 16; ++kt) {
      v16bf av0 = afrag(pt, 512, mt0,     kt, lane);
      v16bf av1 = afrag(pt, 512, mt0 + 1, kt, lane);
      v16bf bv0 = bfrag(u1b, nt0 * 16 + kt, lane);
      v16bf bv1 = bfrag(u1b, (nt0 + 1) * 16 + kt, lane);
      a00 = wmma_bf16(av0, bv0, a00);
      a01 = wmma_bf16(av0, bv1, a01);
      a10 = wmma_bf16(av1, bv0, a10);
      a11 = wmma_bf16(av1, bv1, a11);
    }
#pragma unroll
    for (int j = 0; j < 8; ++j) {
      int r0 = (mt0 * 16 + j + hi8) * 512, r1 = ((mt0 + 1) * 16 + j + hi8) * 512;
      h2[r0 + nt0 * 16 + Nc]       = f2bf(fmaxf(a00[j], 0.0f));
      h2[r0 + (nt0 + 1) * 16 + Nc] = f2bf(fmaxf(a01[j], 0.0f));
      h2[r1 + nt0 * 16 + Nc]       = f2bf(fmaxf(a10[j], 0.0f));
      h2[r1 + (nt0 + 1) * 16 + Nc] = f2bf(fmaxf(a11[j], 0.0f));
    }
  }
  __syncthreads();

  // out = relu(h2 @ u2 + c2) + obj @ pw + pb.  2mtb x 4ntb = 8 blocks.
  for (int tb = w; tb < 8; tb += 8) {
    int nt0 = (tb & 3) * 2, mt0 = (tb >> 2) * 2;
    v8f a00 = bcast8(c2v[nt0 * 16 + Nc]);
    v8f a01 = bcast8(c2v[(nt0 + 1) * 16 + Nc]);
    v8f a10 = a00, a11 = a01;
#pragma unroll
    for (int kt = 0; kt < 16; ++kt) {
      v16bf av0 = afrag(h2, 512, mt0,     kt, lane);
      v16bf av1 = afrag(h2, 512, mt0 + 1, kt, lane);
      v16bf bv0 = bfrag(u2b, nt0 * 16 + kt, lane);
      v16bf bv1 = bfrag(u2b, (nt0 + 1) * 16 + kt, lane);
      a00 = wmma_bf16(av0, bv0, a00);
      a01 = wmma_bf16(av0, bv1, a01);
      a10 = wmma_bf16(av1, bv0, a10);
      a11 = wmma_bf16(av1, bv1, a11);
    }
    // residual: obj @ pw + pb  (K = 128)
    v8f r00 = bcast8(pbv[nt0 * 16 + Nc]);
    v8f r01 = bcast8(pbv[(nt0 + 1) * 16 + Nc]);
    v8f r10 = r00, r11 = r01;
#pragma unroll
    for (int kt = 0; kt < 4; ++kt) {
      v16bf av0 = afrag(ot, 128, mt0,     kt, lane);
      v16bf av1 = afrag(ot, 128, mt0 + 1, kt, lane);
      v16bf bv0 = bfrag(pwb, nt0 * 4 + kt, lane);
      v16bf bv1 = bfrag(pwb, (nt0 + 1) * 4 + kt, lane);
      r00 = wmma_bf16(av0, bv0, r00);
      r01 = wmma_bf16(av0, bv1, r01);
      r10 = wmma_bf16(av1, bv0, r10);
      r11 = wmma_bf16(av1, bv1, r11);
    }
#pragma unroll
    for (int j = 0; j < 8; ++j) {
      int g0 = row0 + mt0 * 16 + j + hi8;
      int g1 = row0 + (mt0 + 1) * 16 + j + hi8;
      if (g0 < NOBJ) {
        outo[(size_t)g0 * DOUTD + nt0 * 16 + Nc]       = fmaxf(a00[j], 0.0f) + r00[j];
        outo[(size_t)g0 * DOUTD + (nt0 + 1) * 16 + Nc] = fmaxf(a01[j], 0.0f) + r01[j];
      }
      if (g1 < NOBJ) {
        outo[(size_t)g1 * DOUTD + nt0 * 16 + Nc]       = fmaxf(a10[j], 0.0f) + r10[j];
        outo[(size_t)g1 * DOUTD + (nt0 + 1) * 16 + Nc] = fmaxf(a11[j], 0.0f) + r11[j];
      }
    }
  }
}

// ---- kernel: pred residual: new_p += pred @ ppw + ppb ----------------------

__global__ __launch_bounds__(256) void predres_kernel(
    const float* __restrict__ pred, const float* __restrict__ ppbv,
    const unsigned short* __restrict__ ppwb, float* __restrict__ outp) {
  extern __shared__ __align__(16) char smem[];
  unsigned short* pt = (unsigned short*)smem;   // 64x256

  int tid  = threadIdx.x;
  int row0 = blockIdx.x * 64;
  {
    int srow  = tid >> 2;
    int cbase = (tid & 3) << 6;
    const float* src = pred + (size_t)(row0 + srow) * DPRED + cbase;
    unsigned short* drow = pt + srow * 256 + cbase;
#pragma unroll
    for (int i = 0; i < 16; ++i)
      st4bf(drow + i * 4, *(const float4*)(src + i * 4), 1.0f);
  }
  __syncthreads();

  int w = tid >> 5, lane = tid & 31;
  int Nc  = lane & 15;
  int hi8 = (lane & 16) >> 1;

  // 2mtb x 4ntb = 8 blocks, one per wave.
  for (int tb = w; tb < 8; tb += 8) {
    int nt0 = (tb & 3) * 2, mt0 = (tb >> 2) * 2;
    v8f a00 = bcast8(ppbv[nt0 * 16 + Nc]);
    v8f a01 = bcast8(ppbv[(nt0 + 1) * 16 + Nc]);
    v8f a10 = a00, a11 = a01;
#pragma unroll
    for (int kt = 0; kt < 8; ++kt) {            // K = 256
      v16bf av0 = afrag(pt, 256, mt0,     kt, lane);
      v16bf av1 = afrag(pt, 256, mt0 + 1, kt, lane);
      v16bf bv0 = bfrag(ppwb, nt0 * 8 + kt, lane);
      v16bf bv1 = bfrag(ppwb, (nt0 + 1) * 8 + kt, lane);
      a00 = wmma_bf16(av0, bv0, a00);
      a01 = wmma_bf16(av0, bv1, a01);
      a10 = wmma_bf16(av1, bv0, a10);
      a11 = wmma_bf16(av1, bv1, a11);
    }
#pragma unroll
    for (int j = 0; j < 8; ++j) {
      size_t r0 = (size_t)(row0 + mt0 * 16 + j + hi8) * DOUTD;
      size_t r1 = (size_t)(row0 + (mt0 + 1) * 16 + j + hi8) * DOUTD;
      outp[r0 + nt0 * 16 + Nc]       += a00[j];
      outp[r0 + (nt0 + 1) * 16 + Nc] += a01[j];
      outp[r1 + nt0 * 16 + Nc]       += a10[j];
      outp[r1 + (nt0 + 1) * 16 + Nc] += a11[j];
    }
  }
}

// ---- launcher --------------------------------------------------------------

extern "C" void kernel_launch(void* const* d_in, const int* in_sizes, int n_in,
                              void* d_out, int out_size, void* d_ws, size_t ws_size,
                              hipStream_t stream) {
  (void)in_sizes; (void)n_in; (void)out_size; (void)ws_size;
  const float* obj  = (const float*)d_in[0];
  const float* pred = (const float*)d_in[1];
  const int*   edges= (const int*)  d_in[2];
  const float* w1   = (const float*)d_in[3];
  const float* b1   = (const float*)d_in[4];
  const float* w2   = (const float*)d_in[5];
  const float* b2   = (const float*)d_in[6];
  const float* u1   = (const float*)d_in[7];
  const float* c1   = (const float*)d_in[8];
  const float* u2   = (const float*)d_in[9];
  const float* c2   = (const float*)d_in[10];
  const float* pw   = (const float*)d_in[11];
  const float* pb   = (const float*)d_in[12];
  const float* ppw  = (const float*)d_in[13];
  const float* ppb  = (const float*)d_in[14];

  float* out  = (float*)d_out;
  float* outo = out;                              // 10000 x 128
  float* outp = out + (size_t)NOBJ * DOUTD;       // 160000 x 128

  // workspace layout
  float* wsf    = (float*)d_ws;
  float* pooled = wsf;                                   // NOBJ_PAD x 512
  float* counts = wsf + (size_t)NOBJ_PAD * HDIM;         // NOBJ_PAD
  unsigned short* wb = (unsigned short*)(counts + NOBJ_PAD);
  unsigned short* w1b  = wb;                // 512x512   -> 262144
  unsigned short* w2b  = wb + 262144;       // 512x1152  -> 589824
  unsigned short* u1b  = wb + 851968;       // 512x512   -> 262144
  unsigned short* u2b  = wb + 1114112;      // 512x128   -> 65536
  unsigned short* pwb  = wb + 1179648;      // 128x128   -> 16384
  unsigned short* ppwb = wb + 1196032;      // 256x128   -> 32768

  int nz = NOBJ_PAD * HDIM + NOBJ_PAD;
  zero_kernel<<<(nz + 255) / 256, 256, 0, stream>>>(wsf, nz);

  prep_kernel<<<(512 * 512 + 255) / 256,  256, 0, stream>>>(w1,  w1b,  512, 512);
  prep_kernel<<<(512 * 1152 + 255) / 256, 256, 0, stream>>>(w2,  w2b,  512, 1152);
  prep_kernel<<<(512 * 512 + 255) / 256,  256, 0, stream>>>(u1,  u1b,  512, 512);
  prep_kernel<<<(512 * 128 + 255) / 256,  256, 0, stream>>>(u2,  u2b,  512, 128);
  prep_kernel<<<(128 * 128 + 255) / 256,  256, 0, stream>>>(pw,  pwb,  128, 128);
  prep_kernel<<<(256 * 128 + 255) / 256,  256, 0, stream>>>(ppw, ppwb, 256, 128);

  count_kernel<<<(NTRI + 255) / 256, 256, 0, stream>>>(edges, counts);

  size_t shT = (size_t)(64 * 512 + 64 * 512) * 2 + 64 * 8;        // 131584
  triples_kernel<<<NTRI / 64, 256, shT, stream>>>(obj, pred, edges, b1, b2,
                                                  w1b, w2b, pooled, outp);

  size_t shP = ((size_t)64 * 512 * 2 + 64 * 128) * 2;             // 147456
  obj_kernel<<<NOBJ_PAD / 64, 256, shP, stream>>>(obj, c1, c2, pb, u1b, u2b,
                                                  pwb, pooled, counts, outo);

  size_t shR = (size_t)64 * 256 * 2;                              // 32768
  predres_kernel<<<NTRI / 64, 256, shR, stream>>>(pred, ppb, ppwb, outp);
}
